// KerasGRULayer_90031104459512
// MI455X (gfx1250) — compile-verified, matrix operand order checked by reading, not run on hardware
//
#include <hip/hip_runtime.h>

// GRU forward for MI455X (gfx1250, wave32, WMMA).
// 4 batch tiles x 16 cooperating WGs; weight slices resident in LDS (f16);
// f16 WMMA (f32 accum) for the fused input projection + recurrent GEMM;
// async global->LDS copy for the h operand; prefetch of next-step inputs;
// software phase barrier + double-buffered h in d_ws.

typedef __attribute__((ext_vector_type(16))) _Float16 v16h;
typedef __attribute__((ext_vector_type(8)))  _Float16 v8h;
typedef __attribute__((ext_vector_type(8)))  float    v8f;

#define BATCH   64
#define TSTEPS  512
#define IDIM    512
#define UNITS   512
#define GDIM    (3 * UNITS)

#define WGS_PER_GROUP 16                       // WGs cooperating on one batch tile
#define NGROUPS       (BATCH / 16)             // 4
#define UNITS_PER_WG  (UNITS / WGS_PER_GROUP)  // 32
#define COLS_PER_WG   (3 * UNITS_PER_WG)       // 96 gate columns per WG
#define NTILES        (COLS_PER_WG / 16)       // 6 N-tiles
#define KSPLIT        2                        // waves per N-tile (split K)
#define NWAVES        (NTILES * KSPLIT)        // 12 waves
#define THREADS       (NWAVES * 32)            // 384
#define KBLK          (IDIM / 32)              // 16 k-blocks of K=32

// LDS sizing
#define B_HALVES   (NTILES * KBLK * 32 * 16)   // 49152 halves = 96KB per matrix
#define A_HALVES   (KBLK * 32 * 16)            // 8192 halves = 16KB per matrix
#define G_FLOATS   (KSPLIT * 16 * COLS_PER_WG) // 3072 floats = 12KB per matrix
#define SMEM_BYTES (2 * B_HALVES * 2 + 2 * A_HALVES * 2 + 2 * G_FLOATS * 4) // 253952

// ISA 16-bit operand swizzle (16x32 A / 32x16 B): element (rc, kl in 0..31)
// lives at lane = rc | (hi<<4), half-slot j, hi=(kl>>3)&1, j=kl-8*hi-8*(kl>=16).
__device__ __forceinline__ int swz_lane(int rc, int kl) {
    return rc | (((kl >> 3) & 1) << 4);
}
__device__ __forceinline__ int swz_j(int kl) {
    int hi = (kl >> 3) & 1;
    return kl - 8 * hi - ((kl >= 16) ? 8 : 0);
}

__device__ __forceinline__ float sigmoidf(float x) {
    return 1.0f / (1.0f + __expf(-x));
}

__global__ __launch_bounds__(THREADS) void gru_persistent_kernel(
    const float* __restrict__ inputs,   // [B, T, I]
    const float* __restrict__ Wk,       // [I, 3U]
    const float* __restrict__ Wr,       // [U, 3U]
    const float* __restrict__ bias,     // [2, 3U]
    float* __restrict__ out,            // [B, T, U]
    _Float16* __restrict__ hbuf,        // [NGROUPS][2][16][U] f16 (zeroed)
    unsigned* __restrict__ ctr)         // [NGROUPS] phase counters (zeroed)
{
    extern __shared__ __align__(32) char smem_raw[];
    _Float16* Bx = (_Float16*)smem_raw;           // kernel slice, swizzled
    _Float16* Br = Bx + B_HALVES;                 // recurrent slice, swizzled
    _Float16* Ax = Br + B_HALVES;                 // x_t tile, swizzled
    _Float16* Ah = Ax + A_HALVES;                 // h_t tile, swizzled
    float*    GX = (float*)(Ah + A_HALVES);       // [KSPLIT][16][96] x-proj partials
    float*    GH = GX + G_FLOATS;                 // [KSPLIT][16][96] rec partials

    const int tid  = threadIdx.x;
    const int grp  = blockIdx.x >> 4;             // batch-tile group 0..3
    const int wgi  = blockIdx.x & (WGS_PER_GROUP - 1);
    const int b0   = grp * 16;                    // batch rows [b0, b0+16)
    const int u0   = wgi * UNITS_PER_WG;          // units [u0, u0+32)
    const int wv   = tid >> 5;                    // wave id
    const int tile = wv >> 1;                     // N-tile 0..5
    const int kh   = wv & 1;                      // K half 0..1
    const int L    = tid & 31;                    // lane

    // ---- Preload weight slices into LDS (once), f32 -> f16, WMMA-swizzled.
    for (int idx = tid; idx < NTILES * 16 * IDIM; idx += THREADS) {
        int nt   = idx / (16 * IDIM);
        int rem  = idx - nt * 16 * IDIM;
        int n    = rem / IDIM;                    // col within tile
        int k    = rem - n * IDIM;                // 0..511
        int gate = nt >> 1;
        int gcol = gate * UNITS + u0 + ((nt & 1) << 4) + n;
        int kb = k >> 5, kl = k & 31;
        int d  = ((nt * KBLK + kb) * 32 + swz_lane(n, kl)) * 16 + swz_j(kl);
        Bx[d] = (_Float16)Wk[(size_t)k * GDIM + gcol];
        Br[d] = (_Float16)Wr[(size_t)k * GDIM + gcol];
    }
    __syncthreads();

    const _Float16* hbase = hbuf + (size_t)grp * (2 * 16 * UNITS);

    for (int t = 0; t < TSTEPS; ++t) {
        // ---- Stage A matrices in 8-half (16B) chunks.
        // Chunk c: m = c>>6, c8 = c&63, K0 = c8*8 (each lane record = two
        // K-contiguous 8-half runs, so 16B chunks map 1:1 to memory runs).
        const _Float16* hsrc = hbase + (size_t)(t & 1) * (16 * UNITS);
        const int NCHUNK = 16 * (IDIM / 8);       // 1024
        for (int c = tid; c < NCHUNK; c += THREADS) {
            int m    = c >> 6;
            int c8   = c & 63;
            int K0   = c8 << 3;
            int kb   = c8 >> 2;
            int lane = m | ((c8 & 1) << 4);
            int jb   = ((c8 >> 1) & 1) << 3;
            int hidx = (kb * 32 + lane) * 16 + jb;
            // x_t: 8 f32 -> 8 f16 -> one ds_store_b128
            const float* src = inputs + (size_t)(b0 + m) * TSTEPS * IDIM +
                               (size_t)t * IDIM + K0;
            float4 f0 = ((const float4*)src)[0];
            float4 f1 = ((const float4*)src)[1];
            v8h hx;
            hx[0] = (_Float16)f0.x; hx[1] = (_Float16)f0.y;
            hx[2] = (_Float16)f0.z; hx[3] = (_Float16)f0.w;
            hx[4] = (_Float16)f1.x; hx[5] = (_Float16)f1.y;
            hx[6] = (_Float16)f1.z; hx[7] = (_Float16)f1.w;
            *(v8h*)(Ax + hidx) = hx;
            // prefetch next timestep's input line into near caches
            if ((c & 3) == 0) {
                const float* pf = src + ((t + 1 < TSTEPS) ? IDIM : 0);
                __builtin_prefetch(pf, 0, 1);
            }
            // h_t: async 16B global -> LDS copy (no VGPR round trip)
            unsigned ldsoff = (unsigned)(uintptr_t)(Ah + hidx);
            unsigned long long ga =
                (unsigned long long)(uintptr_t)(hsrc + m * UNITS + K0);
            asm volatile("global_load_async_to_lds_b128 %0, %1, off"
                         :: "v"(ldsoff), "v"(ga) : "memory");
        }
        asm volatile("s_wait_asynccnt 0" ::: "memory");
        __syncthreads();

        // ---- WMMA: each wave does 8 k-blocks of both GEMMs (K split in 2).
        v8f accx = {};
        v8f acch = {};
        const v16h* axp = (const v16h*)Ax;
        const v16h* ahp = (const v16h*)Ah;
        const v16h* bxp = (const v16h*)(Bx + (size_t)tile * KBLK * 32 * 16);
        const v16h* brp = (const v16h*)(Br + (size_t)tile * KBLK * 32 * 16);
        const int kb0 = kh * (KBLK / KSPLIT);
#pragma unroll
        for (int i = 0; i < KBLK / KSPLIT; ++i) {
            int kb = kb0 + i;
            v16h ax = axp[kb * 32 + L];
            v16h bx = bxp[kb * 32 + L];
            accx = __builtin_amdgcn_wmma_f32_16x16x32_f16(
                false, ax, false, bx, (short)0, accx, false, false);
            v16h ah = ahp[kb * 32 + L];
            v16h bh = brp[kb * 32 + L];
            acch = __builtin_amdgcn_wmma_f32_16x16x32_f16(
                false, ah, false, bh, (short)0, acch, false, false);
        }
        // C/D layout: VGPR r holds (M = r + 8*(lane>=16), N = lane&15)
        {
            int col   = tile * 16 + (L & 15);
            int mbase = (L >> 4) * 8;
            float* gx = GX + kh * 16 * COLS_PER_WG;
            float* gh = GH + kh * 16 * COLS_PER_WG;
#pragma unroll
            for (int r = 0; r < 8; ++r) {
                gx[(mbase + r) * COLS_PER_WG + col] = accx[r];
                gh[(mbase + r) * COLS_PER_WG + col] = acch[r];
            }
        }
        __syncthreads();

        // ---- Gate math for this WG's 16x32 h_new tile (sum K partials).
        _Float16* hdst = (_Float16*)hbase + (size_t)((t + 1) & 1) * (16 * UNITS);
        for (int idx = tid; idx < 16 * UNITS_PER_WG; idx += THREADS) {
            int m = idx >> 5, u = idx & (UNITS_PER_WG - 1);
            int gu = u0 + u;
            const float* gx0 = GX + m * COLS_PER_WG;
            const float* gx1 = gx0 + 16 * COLS_PER_WG;
            const float* gh0 = GH + m * COLS_PER_WG;
            const float* gh1 = gh0 + 16 * COLS_PER_WG;
            float xz = gx0[u]      + gx1[u]      + bias[gu];
            float xr = gx0[32 + u] + gx1[32 + u] + bias[UNITS + gu];
            float xh = gx0[64 + u] + gx1[64 + u] + bias[2 * UNITS + gu];
            float rz = gh0[u]      + gh1[u]      + bias[GDIM + gu];
            float rr = gh0[32 + u] + gh1[32 + u] + bias[GDIM + UNITS + gu];
            float rh = gh0[64 + u] + gh1[64 + u] + bias[GDIM + 2 * UNITS + gu];
            // previous h: reuse the swizzled Ah staging (K index == unit)
            int kb = gu >> 5, kl = gu & 31;
            float hp = (float)Ah[(kb * 32 + swz_lane(m, kl)) * 16 + swz_j(kl)];
            float z  = sigmoidf(xz + rz);
            float r  = sigmoidf(xr + rr);
            float hh = tanhf(xh + r * rh);
            float hn = z * hp + (1.0f - z) * hh;
            out[(size_t)(b0 + m) * TSTEPS * UNITS + (size_t)t * UNITS + gu] = hn;
            hdst[m * UNITS + gu] = (_Float16)hn;
        }

        // ---- Software phase barrier across the 16 WGs of this group.
        __threadfence();          // release h_new to device scope
        __syncthreads();
        if (tid == 0) {
            __hip_atomic_fetch_add(&ctr[grp], 1u, __ATOMIC_RELEASE,
                                   __HIP_MEMORY_SCOPE_AGENT);
            unsigned target = (unsigned)(WGS_PER_GROUP * (t + 1));
            while (__hip_atomic_load(&ctr[grp], __ATOMIC_ACQUIRE,
                                     __HIP_MEMORY_SCOPE_AGENT) < target) {
                __builtin_amdgcn_s_sleep(2);
            }
        }
        __syncthreads();
        __threadfence();          // acquire: invalidate stale near caches
    }
}

extern "C" void kernel_launch(void* const* d_in, const int* in_sizes, int n_in,
                              void* d_out, int out_size, void* d_ws, size_t ws_size,
                              hipStream_t stream) {
    const float* inputs = (const float*)d_in[0];
    const float* Wk     = (const float*)d_in[1];
    const float* Wr     = (const float*)d_in[2];
    const float* bias   = (const float*)d_in[3];
    float* out          = (float*)d_out;

    // Workspace: [0,256) phase counters; [256, 256+128KB) h double buffers.
    const size_t HBUF_BYTES = (size_t)NGROUPS * 2 * 16 * UNITS * sizeof(_Float16);
    unsigned* ctr   = (unsigned*)d_ws;
    _Float16* hbuf  = (_Float16*)((char*)d_ws + 256);
    hipMemsetAsync(d_ws, 0, 256 + HBUF_BYTES, stream);   // h0 = 0, counters = 0

    hipFuncSetAttribute((const void*)gru_persistent_kernel,
                        hipFuncAttributeMaxDynamicSharedMemorySize, SMEM_BYTES);

    dim3 grid(NGROUPS * WGS_PER_GROUP);   // 64 persistent workgroups
    dim3 block(THREADS);                  // 12 waves (wave32), 3 per SIMD32
    gru_persistent_kernel<<<grid, block, SMEM_BYTES, stream>>>(
        inputs, Wk, Wr, bias, out, hbuf, ctr);
}
